// model_fn_63874753626130
// MI455X (gfx1250) — compile-verified, matrix operand order checked by reading, not run on hardware
//
#include <hip/hip_runtime.h>
#include <cstdint>

// ---------------------------------------------------------------------------
// Types / helpers
// ---------------------------------------------------------------------------
typedef __attribute__((ext_vector_type(16))) __bf16 bf16x16;
typedef __attribute__((ext_vector_type(8)))  __bf16 bf16x8;
typedef __attribute__((ext_vector_type(8)))  float  f32x8;

#define DEV __device__ __forceinline__

DEV __bf16 f2bf(float f) {
  union { float f; uint32_t u; } a; a.f = f;
  uint32_t u = a.u + 0x7fffu + ((a.u >> 16) & 1u);   // round-to-nearest-even
  union { uint16_t s; __bf16 b; } r; r.s = (uint16_t)(u >> 16);
  return r.b;
}
DEV float sigf(float x) { return 1.0f / (1.0f + expf(-x)); }

// Load a 16x32 bf16 fragment (A or B operand) for v_wmma_f32_16x16x32_bf16.
// Per ISA layout: lane L<16 holds row (L&15), K in [kh*8,kh*8+8) and [16+kh*8,...)
DEV bf16x16 load_frag(const __bf16* base, int ld) {
  int lane = threadIdx.x & 31;
  const __bf16* r = base + (size_t)(lane & 15) * ld + ((lane >> 4) & 1) * 8;
  bf16x8 lo = *(const bf16x8*)(r);
  bf16x8 hi = *(const bf16x8*)(r + 16);
  bf16x16 f;
#pragma unroll
  for (int i = 0; i < 8; ++i) { f[i] = lo[i]; f[i + 8] = hi[i]; }
  return f;
}

DEV f32x8 wmma_bf16(bf16x16 a, bf16x16 b, f32x8 c) {
  return __builtin_amdgcn_wmma_f32_16x16x32_bf16(false, a, false, b, (short)0, c,
                                                 false, false);
}

// ---------------------------------------------------------------------------
// Problem constants
// ---------------------------------------------------------------------------
static constexpr int BB  = 16;
static constexpr int TQ  = 128;
static constexpr int TR  = 256;
static constexpr int H   = 512;   // FEAT == MEM == ATT == 512

// ---------------------------------------------------------------------------
// Workspace layout (bytes). All blocks are multiples of 16KB -> aligned.
// ZPRE_PTR aliases ZPRE_R (encoder-r precompute dead by pointer-LSTM stage).
// ---------------------------------------------------------------------------
static constexpr size_t O_CIH   = 0;                                   // cell Wih bf16 2048x512
static constexpr size_t O_CHH   = O_CIH   + 2048ull*512*2;
static constexpr size_t O_FIH   = O_CHH   + 2048ull*512*2;             // fwd cell
static constexpr size_t O_FHH   = O_FIH   + 2048ull*512*2;
static constexpr size_t O_BIH   = O_FHH   + 2048ull*512*2;             // bwd cell
static constexpr size_t O_BHH   = O_BIH   + 2048ull*512*2;
static constexpr size_t O_PIH   = O_BHH   + 2048ull*512*2;             // ptr Wih 2048x1024
static constexpr size_t O_PHH   = O_PIH   + 2048ull*1024*2;
static constexpr size_t O_M1WQ  = O_PHH   + 2048ull*512*2;
static constexpr size_t O_M1WPX = O_M1WQ  + 512ull*512*2;
static constexpr size_t O_M1WPH = O_M1WPX + 512ull*512*2;
static constexpr size_t O_M1WIN = O_M1WPH + 512ull*512*2;
static constexpr size_t O_M2WQ  = O_M1WIN + 512ull*512*2;
static constexpr size_t O_M2WPX = O_M2WQ  + 512ull*512*2;
static constexpr size_t O_M2WPH = O_M2WPX + 512ull*512*2;
static constexpr size_t O_M2WIN = O_M2WPH + 512ull*512*2;
static constexpr size_t O_QBF   = O_M2WIN + 512ull*512*2;              // query bf16
static constexpr size_t O_RBF   = O_QBF   + (size_t)BB*TQ*H*2;         // ref bf16
static constexpr size_t O_ZPREQ = O_RBF   + (size_t)BB*TR*H*2;         // q @ Wih^T  f32
static constexpr size_t O_ZPRER = O_ZPREQ + (size_t)BB*TQ*2048*4;      // r @ Wih^T  f32 (alias ptr zpre)
static constexpr size_t O_HQF   = O_ZPRER + (size_t)BB*TR*2048*4;      // hq f32
static constexpr size_t O_HQBF  = O_HQF   + (size_t)BB*TQ*H*4;
static constexpr size_t O_HRF   = O_HQBF  + (size_t)BB*TQ*H*2;         // hr f32
static constexpr size_t O_HRBF  = O_HRF   + (size_t)BB*TR*H*4;
static constexpr size_t O_HRRF  = O_HRBF  + (size_t)BB*TR*H*2;         // hr reversed f32
static constexpr size_t O_HRRBF = O_HRRF  + (size_t)BB*TR*H*4;
static constexpr size_t O_HQP1  = O_HRRBF + (size_t)BB*TR*H*2;         // hq @ Wq1^T
static constexpr size_t O_HQP2  = O_HQP1  + (size_t)BB*TQ*H*4;
static constexpr size_t O_HPXF  = O_HQP2  + (size_t)BB*TQ*H*4;         // hr @ WpX1^T
static constexpr size_t O_HPXB  = O_HPXF  + (size_t)BB*TR*H*4;         // hrrev @ WpX2^T
static constexpr size_t O_MATCH = O_HPXB  + (size_t)BB*TR*H*4;         // match bf16 (B,TR,1024)
static constexpr size_t O_PTRH  = O_MATCH + (size_t)BB*TR*1024*2;      // ptr hidden seq f32
static constexpr size_t O_H0    = O_PTRH  + (size_t)BB*TR*H*4;         // h state bf16 (double buf)
static constexpr size_t O_H1    = O_H0    + (size_t)BB*H*2;
static constexpr size_t O_CST   = O_H1    + (size_t)BB*H*2;            // c state f32
static constexpr size_t O_HP    = O_CST   + (size_t)BB*H*4;            // hp f32 16x512
static constexpr size_t O_AVF   = O_HP    + (size_t)BB*H*4;            // av f32
static constexpr size_t O_AVBF  = O_AVF   + (size_t)BB*H*4;            // av bf16
static constexpr size_t O_IN1   = O_AVBF  + (size_t)BB*H*2;            // in1 bf16

// ---------------------------------------------------------------------------
// Kernels
// ---------------------------------------------------------------------------
__global__ void k_zero(uint32_t* p, int n) {
  int i = blockIdx.x * blockDim.x + threadIdx.x;
  if (i < n) p[i] = 0u;
}

// dst[r*cols+c] = bf16(src[r*stride + off + c])
__global__ void k_convert(const float* __restrict__ src, __bf16* __restrict__ dst,
                          int total, int cols, int stride, int off) {
  int i = blockIdx.x * blockDim.x + threadIdx.x;
  if (i >= total) return;
  int r = i / cols, c = i - r * cols;
  dst[i] = f2bf(src[(size_t)r * stride + off + c]);
}

// C[M,N] = A[M,K] @ W[N,K]^T   (bf16 in, f32 out). grid=(N/16, M/64), block=128.
__global__ void k_gemm_bf16(const __bf16* __restrict__ A, const __bf16* __restrict__ W,
                            float* __restrict__ C, int N, int K) {
  int wave = threadIdx.x >> 5;
  int mt = blockIdx.y * 4 + wave;
  int nt = blockIdx.x;
  const __bf16* Ab = A + (size_t)mt * 16 * K;
  const __bf16* Wb = W + (size_t)nt * 16 * K;
  f32x8 acc = {};
  for (int k0 = 0; k0 < K; k0 += 32)
    acc = wmma_bf16(load_frag(Ab + k0, K), load_frag(Wb + k0, K), acc);
  int lane = threadIdx.x & 31;
  int n = lane & 15, mh = (lane >> 4) & 1;
#pragma unroll
  for (int r = 0; r < 8; ++r)
    C[((size_t)mt * 16 + r + 8 * mh) * N + nt * 16 + n] = acc[r];
}

// One LSTM step with precomputed input contribution.
// z = zpre(step t) + h @ Whh^T + bih + bhh -> gates -> c,h update (+mask)
// grid=32 (16 hidden cols each), block=128 (wave g = gate block g).
__global__ void k_lstm_step(const float* __restrict__ zpre, int zRowStride,
                            const __bf16* __restrict__ hin, const __bf16* __restrict__ Whh,
                            const float* __restrict__ bih, const float* __restrict__ bhh,
                            __bf16* __restrict__ hout, float* __restrict__ c,
                            const int* __restrict__ len,
                            float* __restrict__ outF, __bf16* __restrict__ outB,
                            int T, int t) {
  __shared__ float zb[4][16][16];
  int wave = threadIdx.x >> 5, lane = threadIdx.x & 31;
  int j0 = blockIdx.x * 16;
  int n0 = wave * 512 + j0;
  int nn = lane & 15, mh = (lane >> 4) & 1;
  f32x8 acc;
#pragma unroll
  for (int r = 0; r < 8; ++r)
    acc[r] = zpre[(size_t)(r + 8 * mh) * zRowStride + n0 + nn];
  for (int k0 = 0; k0 < 512; k0 += 32)
    acc = wmma_bf16(load_frag(hin + k0, 512),
                    load_frag(Whh + (size_t)n0 * 512 + k0, 512), acc);
#pragma unroll
  for (int r = 0; r < 8; ++r) zb[wave][r + 8 * mh][nn] = acc[r];
  __syncthreads();
  for (int e = threadIdx.x; e < 256; e += 128) {
    int b = e >> 4, jj = e & 15, j = j0 + jj;
    float zi = zb[0][b][jj] + bih[j]        + bhh[j];
    float zf = zb[1][b][jj] + bih[512 + j]  + bhh[512 + j];
    float zg = zb[2][b][jj] + bih[1024 + j] + bhh[1024 + j];
    float zo = zb[3][b][jj] + bih[1536 + j] + bhh[1536 + j];
    float cn = sigf(zf) * c[b * 512 + j] + sigf(zi) * tanhf(zg);
    float hn = sigf(zo) * tanhf(cn);
    if (len && j >= len[b]) { cn = 0.f; hn = 0.f; }    // dynamic_rnn dim mask
    c[b * 512 + j] = cn;
    hout[b * 512 + j] = f2bf(hn);
    if (outF) outF[((size_t)b * T + t) * 512 + j] = hn;
    if (outB) outB[((size_t)b * T + t) * 512 + j] = f2bf(hn);
  }
}

// Match-cell LSTM step: z = in1 @ Wih^T + h @ Whh^T + biases -> update,
// write h into match buffer at (possibly flipped) position.
__global__ void k_match_lstm_step(const __bf16* __restrict__ in1, const __bf16* __restrict__ Wih,
                                  const __bf16* __restrict__ hin, const __bf16* __restrict__ Whh,
                                  const float* __restrict__ bih, const float* __restrict__ bhh,
                                  __bf16* __restrict__ hout, float* __restrict__ c,
                                  __bf16* __restrict__ matchOut, int colOff,
                                  const int* __restrict__ flipLen, int t) {
  __shared__ float zb[4][16][16];
  int wave = threadIdx.x >> 5, lane = threadIdx.x & 31;
  int j0 = blockIdx.x * 16;
  int n0 = wave * 512 + j0;
  f32x8 acc = {};
  for (int k0 = 0; k0 < 512; k0 += 32)
    acc = wmma_bf16(load_frag(in1 + k0, 512),
                    load_frag(Wih + (size_t)n0 * 512 + k0, 512), acc);
  for (int k0 = 0; k0 < 512; k0 += 32)
    acc = wmma_bf16(load_frag(hin + k0, 512),
                    load_frag(Whh + (size_t)n0 * 512 + k0, 512), acc);
  int nn = lane & 15, mh = (lane >> 4) & 1;
#pragma unroll
  for (int r = 0; r < 8; ++r) zb[wave][r + 8 * mh][nn] = acc[r];
  __syncthreads();
  for (int e = threadIdx.x; e < 256; e += 128) {
    int b = e >> 4, jj = e & 15, j = j0 + jj;
    float zi = zb[0][b][jj] + bih[j]        + bhh[j];
    float zf = zb[1][b][jj] + bih[512 + j]  + bhh[512 + j];
    float zg = zb[2][b][jj] + bih[1024 + j] + bhh[1024 + j];
    float zo = zb[3][b][jj] + bih[1536 + j] + bhh[1536 + j];
    float cn = sigf(zf) * c[b * 512 + j] + sigf(zi) * tanhf(zg);
    float hn = sigf(zo) * tanhf(cn);
    c[b * 512 + j] = cn;
    hout[b * 512 + j] = f2bf(hn);
    int td = t;
    if (flipLen) { int L = flipLen[b]; td = (t < L) ? (L - 1 - t) : t; }
    matchOut[((size_t)b * TR + td) * 1024 + colOff + j] = f2bf(hn);
  }
}

// hp = hpx[:,t,:] + bp + h @ WpH^T    (16x512). grid=8, block=128.
__global__ void k_hp(const __bf16* __restrict__ hin, const __bf16* __restrict__ WpH,
                     const float* __restrict__ hpx, const float* __restrict__ bp,
                     float* __restrict__ hp, int t) {
  int wave = threadIdx.x >> 5;
  int nt = blockIdx.x * 4 + wave;
  f32x8 acc = {};
  for (int k0 = 0; k0 < 512; k0 += 32)
    acc = wmma_bf16(load_frag(hin + k0, 512),
                    load_frag(WpH + (size_t)nt * 16 * 512 + k0, 512), acc);
  int lane = threadIdx.x & 31, n = lane & 15, mh = (lane >> 4) & 1;
#pragma unroll
  for (int r = 0; r < 8; ++r) {
    int m = r + 8 * mh, col = nt * 16 + n;
    hp[m * 512 + col] = acc[r] + hpx[((size_t)m * TR + t) * 512 + col] + bp[col];
  }
}

// Attention scores + masked softmax + av. grid=16 (per batch), block=128 (==TQ).
__global__ void k_attn_av(const float* __restrict__ hq_proj, const float* __restrict__ hp,
                          const float* __restrict__ wg, const float* __restrict__ bg,
                          const float* __restrict__ hq, const int* __restrict__ len_q,
                          float* __restrict__ av_f, __bf16* __restrict__ av_bf) {
  __shared__ float sal[128];
  __shared__ float red[128];
  int b = blockIdx.x, tid = threadIdx.x;
  const float* hqp = hq_proj + ((size_t)b * TQ + tid) * 512;
  const float* hpb = hp + b * 512;
  float acc = 0.f;
  for (int k = 0; k < 512; ++k) acc += wg[k] * tanhf(hqp[k] + hpb[k]);
  float s = acc + bg[0];
  if (tid >= len_q[b]) s = -1e9f;
  red[tid] = s; __syncthreads();
  for (int st = 64; st > 0; st >>= 1) {
    if (tid < st) red[tid] = fmaxf(red[tid], red[tid + st]);
    __syncthreads();
  }
  float mx = red[0]; __syncthreads();
  float e = expf(s - mx);
  red[tid] = e; __syncthreads();
  for (int st = 64; st > 0; st >>= 1) {
    if (tid < st) red[tid] += red[tid + st];
    __syncthreads();
  }
  float inv = 1.0f / red[0];
  sal[tid] = e * inv; __syncthreads();
  for (int m = tid; m < 512; m += 128) {
    float a = 0.f;
    for (int t = 0; t < TQ; ++t) a += sal[t] * hq[((size_t)b * TQ + t) * 512 + m];
    av_f[b * 512 + m] = a;
    av_bf[b * 512 + m] = f2bf(a);
  }
}

// in1 = x_t * sigmoid(av @ Win^T + bin). grid=8, block=128.
__global__ void k_gate_in1(const __bf16* __restrict__ av, const __bf16* __restrict__ Win,
                           const float* __restrict__ bin, const float* __restrict__ xseq,
                           __bf16* __restrict__ in1, int t) {
  int wave = threadIdx.x >> 5;
  int nt = blockIdx.x * 4 + wave;
  f32x8 acc = {};
  for (int k0 = 0; k0 < 512; k0 += 32)
    acc = wmma_bf16(load_frag(av + k0, 512),
                    load_frag(Win + (size_t)nt * 16 * 512 + k0, 512), acc);
  int lane = threadIdx.x & 31, n = lane & 15, mh = (lane >> 4) & 1;
#pragma unroll
  for (int r = 0; r < 8; ++r) {
    int m = r + 8 * mh, col = nt * 16 + n;
    float g = sigf(acc[r] + bin[col]);
    in1[m * 512 + col] = f2bf(xseq[((size_t)m * TR + t) * 512 + col] * g);
  }
}

// flipBatch: reverse first len[b] steps. idx over 16*256*512.
__global__ void k_flip(const float* __restrict__ src, float* __restrict__ dstF,
                       __bf16* __restrict__ dstB, const int* __restrict__ len) {
  int idx = blockIdx.x * blockDim.x + threadIdx.x;
  if (idx >= BB * TR * 512) return;
  int j = idx & 511;
  int t = (idx >> 9) & (TR - 1);
  int b = idx >> 17;
  int L = len[b];
  int st = (t < L) ? (L - 1 - t) : t;
  float v = src[((size_t)b * TR + st) * 512 + j];
  dstF[idx] = v;
  dstB[idx] = f2bf(v);
}

// out = ptr_h @ fc_w^T + fc_b    (16*256 x 4)
__global__ void k_fc(const float* __restrict__ h, const float* __restrict__ w,
                     const float* __restrict__ bias, float* __restrict__ out) {
  int idx = blockIdx.x * blockDim.x + threadIdx.x;
  if (idx >= BB * TR * 4) return;
  int j = idx & 3, row = idx >> 2;
  const float* hr = h + (size_t)row * 512;
  const float* wr = w + j * 512;
  float a = bias[j];
  for (int k = 0; k < 512; ++k) a += hr[k] * wr[k];
  out[idx] = a;
}

// ---------------------------------------------------------------------------
// Host orchestration
// ---------------------------------------------------------------------------
extern "C" void kernel_launch(void* const* d_in, const int* in_sizes, int n_in,
                              void* d_out, int out_size, void* d_ws, size_t ws_size,
                              hipStream_t stream) {
  (void)in_sizes; (void)n_in; (void)out_size; (void)ws_size;
  const float* query = (const float*)d_in[0];
  const float* refp  = (const float*)d_in[1];
  const int* len_q   = (const int*)d_in[2];
  const int* len_r   = (const int*)d_in[3];
  int p = 4;
  // cell
  const float* cWih = (const float*)d_in[p++]; const float* cWhh = (const float*)d_in[p++];
  const float* cbih = (const float*)d_in[p++]; const float* cbhh = (const float*)d_in[p++];
  // fwd cell
  const float* fWih = (const float*)d_in[p++]; const float* fWhh = (const float*)d_in[p++];
  const float* fbih = (const float*)d_in[p++]; const float* fbhh = (const float*)d_in[p++];
  // bwd cell
  const float* bWih = (const float*)d_in[p++]; const float* bWhh = (const float*)d_in[p++];
  const float* bbih = (const float*)d_in[p++]; const float* bbhh = (const float*)d_in[p++];
  // m1: Wq, Wp, bp, wg, bg, Whq, bhq, Win, bin, Wx, bx, Wy, by
  const float* m1Wq = (const float*)d_in[p++]; const float* m1Wp = (const float*)d_in[p++];
  const float* m1bp = (const float*)d_in[p++]; const float* m1wg = (const float*)d_in[p++];
  const float* m1bg = (const float*)d_in[p++]; p += 2;  // Whq, bhq dead
  const float* m1Win = (const float*)d_in[p++]; const float* m1bin = (const float*)d_in[p++];
  p += 4;  // Wx, bx, Wy, by dead
  // m2
  const float* m2Wq = (const float*)d_in[p++]; const float* m2Wp = (const float*)d_in[p++];
  const float* m2bp = (const float*)d_in[p++]; const float* m2wg = (const float*)d_in[p++];
  const float* m2bg = (const float*)d_in[p++]; p += 2;
  const float* m2Win = (const float*)d_in[p++]; const float* m2bin = (const float*)d_in[p++];
  p += 4;
  // ptr cell
  const float* pWih = (const float*)d_in[p++]; const float* pWhh = (const float*)d_in[p++];
  const float* pbih = (const float*)d_in[p++]; const float* pbhh = (const float*)d_in[p++];
  const float* fc_w = (const float*)d_in[p++]; const float* fc_b = (const float*)d_in[p++];

  char* ws = (char*)d_ws;
  auto BF = [&](size_t o) { return (__bf16*)(ws + o); };
  auto FP = [&](size_t o) { return (float*)(ws + o); };

  const int TPB = 256;
  auto cdiv = [](long a, long b) { return (int)((a + b - 1) / b); };
  auto conv = [&](const float* s, size_t dstOff, int rows, int cols, int stride, int off) {
    k_convert<<<cdiv((long)rows * cols, TPB), TPB, 0, stream>>>(s, BF(dstOff),
                                                               rows * cols, cols, stride, off);
  };
  auto zero = [&](size_t off, size_t bytes) {
    k_zero<<<cdiv((long)(bytes / 4), TPB), TPB, 0, stream>>>((uint32_t*)(ws + off),
                                                             (int)(bytes / 4));
  };
  auto gemm = [&](size_t aOff, size_t wOff, size_t cOff, int M, int N, int K) {
    dim3 g(N / 16, M / 64);
    k_gemm_bf16<<<g, 128, 0, stream>>>(BF(aOff), BF(wOff), FP(cOff), N, K);
  };

  // ---- 1. convert all weights + activations to bf16 ----
  conv(cWih, O_CIH, 2048, 512, 512, 0);   conv(cWhh, O_CHH, 2048, 512, 512, 0);
  conv(fWih, O_FIH, 2048, 512, 512, 0);   conv(fWhh, O_FHH, 2048, 512, 512, 0);
  conv(bWih, O_BIH, 2048, 512, 512, 0);   conv(bWhh, O_BHH, 2048, 512, 512, 0);
  conv(pWih, O_PIH, 2048, 1024, 1024, 0); conv(pWhh, O_PHH, 2048, 512, 512, 0);
  conv(m1Wq, O_M1WQ, 512, 512, 512, 0);
  conv(m1Wp, O_M1WPX, 512, 512, 1024, 0);   conv(m1Wp, O_M1WPH, 512, 512, 1024, 512);
  conv(m1Win, O_M1WIN, 512, 512, 512, 0);
  conv(m2Wq, O_M2WQ, 512, 512, 512, 0);
  conv(m2Wp, O_M2WPX, 512, 512, 1024, 0);   conv(m2Wp, O_M2WPH, 512, 512, 1024, 512);
  conv(m2Win, O_M2WIN, 512, 512, 512, 0);
  conv(query, O_QBF, BB * TQ, 512, 512, 0);
  conv(refp,  O_RBF, BB * TR, 512, 512, 0);

  // ---- 2. big precompute GEMMs: x @ Wih^T for both encoder passes ----
  gemm(O_QBF, O_CIH, O_ZPREQ, BB * TQ, 2048, 512);
  gemm(O_RBF, O_CIH, O_ZPRER, BB * TR, 2048, 512);

  // ---- 3. encoder over query ----
  zero(O_H0, (size_t)BB * H * 2); zero(O_CST, (size_t)BB * H * 4);
  for (int t = 0; t < TQ; ++t) {
    __bf16* hin = (t & 1) ? BF(O_H1) : BF(O_H0);
    __bf16* hout = (t & 1) ? BF(O_H0) : BF(O_H1);
    k_lstm_step<<<32, 128, 0, stream>>>(FP(O_ZPREQ) + (size_t)t * 2048, TQ * 2048,
                                        hin, BF(O_CHH), cbih, cbhh, hout, FP(O_CST),
                                        len_q, FP(O_HQF), BF(O_HQBF), TQ, t);
  }
  // ---- 4. encoder over ref ----
  zero(O_H0, (size_t)BB * H * 2); zero(O_CST, (size_t)BB * H * 4);
  for (int t = 0; t < TR; ++t) {
    __bf16* hin = (t & 1) ? BF(O_H1) : BF(O_H0);
    __bf16* hout = (t & 1) ? BF(O_H0) : BF(O_H1);
    k_lstm_step<<<32, 128, 0, stream>>>(FP(O_ZPRER) + (size_t)t * 2048, TR * 2048,
                                        hin, BF(O_CHH), cbih, cbhh, hout, FP(O_CST),
                                        len_r, FP(O_HRF), BF(O_HRBF), TR, t);
  }

  // ---- 5. attention precomputes ----
  gemm(O_HQBF, O_M1WQ, O_HQP1, BB * TQ, 512, 512);
  gemm(O_HQBF, O_M2WQ, O_HQP2, BB * TQ, 512, 512);
  gemm(O_HRBF, O_M1WPX, O_HPXF, BB * TR, 512, 512);
  k_flip<<<cdiv((long)BB * TR * 512, TPB), TPB, 0, stream>>>(FP(O_HRF), FP(O_HRRF),
                                                             BF(O_HRRBF), len_r);
  gemm(O_HRRBF, O_M2WPX, O_HPXB, BB * TR, 512, 512);

  // ---- 6. forward match RNN ----
  zero(O_H0, (size_t)BB * H * 2); zero(O_CST, (size_t)BB * H * 4);
  for (int t = 0; t < TR; ++t) {
    __bf16* hin = (t & 1) ? BF(O_H1) : BF(O_H0);
    __bf16* hout = (t & 1) ? BF(O_H0) : BF(O_H1);
    k_hp<<<8, 128, 0, stream>>>(hin, BF(O_M1WPH), FP(O_HPXF), m1bp, FP(O_HP), t);
    k_attn_av<<<16, 128, 0, stream>>>(FP(O_HQP1), FP(O_HP), m1wg, m1bg, FP(O_HQF),
                                      len_q, FP(O_AVF), BF(O_AVBF));
    k_gate_in1<<<8, 128, 0, stream>>>(BF(O_AVBF), BF(O_M1WIN), m1bin, FP(O_HRF),
                                      BF(O_IN1), t);
    k_match_lstm_step<<<32, 128, 0, stream>>>(BF(O_IN1), BF(O_FIH), hin, BF(O_FHH),
                                              fbih, fbhh, hout, FP(O_CST),
                                              BF(O_MATCH), 0, nullptr, t);
  }
  // ---- 7. backward match RNN (consumes reversed hr; writes flipped) ----
  zero(O_H0, (size_t)BB * H * 2); zero(O_CST, (size_t)BB * H * 4);
  for (int t = 0; t < TR; ++t) {
    __bf16* hin = (t & 1) ? BF(O_H1) : BF(O_H0);
    __bf16* hout = (t & 1) ? BF(O_H0) : BF(O_H1);
    k_hp<<<8, 128, 0, stream>>>(hin, BF(O_M2WPH), FP(O_HPXB), m2bp, FP(O_HP), t);
    k_attn_av<<<16, 128, 0, stream>>>(FP(O_HQP2), FP(O_HP), m2wg, m2bg, FP(O_HQF),
                                      len_q, FP(O_AVF), BF(O_AVBF));
    k_gate_in1<<<8, 128, 0, stream>>>(BF(O_AVBF), BF(O_M2WIN), m2bin, FP(O_HRRF),
                                      BF(O_IN1), t);
    k_match_lstm_step<<<32, 128, 0, stream>>>(BF(O_IN1), BF(O_BIH), hin, BF(O_BHH),
                                              bbih, bbhh, hout, FP(O_CST),
                                              BF(O_MATCH), 512, len_r, t);
  }

  // ---- 8. pointer LSTM: big input precompute, then recurrent steps ----
  gemm(O_MATCH, O_PIH, O_ZPRER /* aliased as zpre_ptr */, BB * TR, 2048, 1024);
  zero(O_H0, (size_t)BB * H * 2); zero(O_CST, (size_t)BB * H * 4);
  for (int t = 0; t < TR; ++t) {
    __bf16* hin = (t & 1) ? BF(O_H1) : BF(O_H0);
    __bf16* hout = (t & 1) ? BF(O_H0) : BF(O_H1);
    k_lstm_step<<<32, 128, 0, stream>>>(FP(O_ZPRER) + (size_t)t * 2048, TR * 2048,
                                        hin, BF(O_PHH), pbih, pbhh, hout, FP(O_CST),
                                        len_r, FP(O_PTRH), nullptr, TR, t);
  }

  // ---- 9. final FC head ----
  k_fc<<<cdiv((long)BB * TR * 4, TPB), TPB, 0, stream>>>(FP(O_PTRH), fc_w, fc_b,
                                                         (float*)d_out);
}
